// Attention_50689204027783
// MI455X (gfx1250) — matrix operand order repaired and checked by reading, not compiled, not run on hardware
//
#include <hip/hip_runtime.h>
#include <hip/hip_bf16.h>
#include <math.h>

// ---------------------------------------------------------------------------
// Types for CDNA5 WMMA (wave32, 16x16x32 bf16 -> f32)
// ---------------------------------------------------------------------------
typedef __bf16 bf16_t;
typedef bf16_t v8bf  __attribute__((ext_vector_type(8)));
typedef bf16_t v16bf __attribute__((ext_vector_type(16)));
typedef float  v8f   __attribute__((ext_vector_type(8)));

#define B_DIM 32
#define T_DIM 2048
#define D_DIM 1024
#define U_DIM 1024
#define M_TILE 64            // rows of B*T per workgroup
#define LDA (D_DIM + 8)      // LDS A row pitch in bf16 elements (bank spread)

// gfx1250 has a hardware V_TANH_F32 transcendental; use it if the builtin
// exists, else a branch-free exp2/rcp identity (saturates correctly to +-1).
#if __has_builtin(__builtin_amdgcn_tanhf)
__device__ __forceinline__ float fast_tanh(float x) { return __builtin_amdgcn_tanhf(x); }
#else
__device__ __forceinline__ float fast_tanh(float x) {
    // tanh(x) = 1 - 2 / (1 + exp2(x * 2*log2(e)))
    float e = __builtin_amdgcn_exp2f(x * 2.8853900817779268f);
    return 1.0f - 2.0f * __builtin_amdgcn_rcpf(1.0f + e);
}
#endif

__device__ __forceinline__ unsigned short f2bf(float f) {
    unsigned u = __builtin_bit_cast(unsigned, f);
    unsigned r = u + 0x7FFFu + ((u >> 16) & 1u);   // round-to-nearest-even
    return (unsigned short)(r >> 16);
}

__device__ __forceinline__ v16bf ld_b(const unsigned short* p) {
    return *(const v16bf*)(const void*)p;
}

// ---------------------------------------------------------------------------
// K0: pack W1 [D,U] f32 row-major -> bf16 layout [kb][u][kk] (kb=K/32, kk=K%32)
// so a wave's B fragment (lane n: col, K half 0..15 / 16..31) is one
// contiguous 32-byte, 32B-aligned load; a full wave covers a contiguous 1KB.
// ---------------------------------------------------------------------------
__global__ void __launch_bounds__(256) pack_w1(const float* __restrict__ W1,
                                               unsigned short* __restrict__ dst) {
    const int kb = blockIdx.x >> 2;                       // 0..31
    const int u  = (blockIdx.x & 3) * 256 + threadIdx.x;  // 0..1023
    unsigned short tmp[32];
#pragma unroll
    for (int kk = 0; kk < 32; ++kk)
        tmp[kk] = f2bf(W1[(size_t)(kb * 32 + kk) * U_DIM + u]);
    unsigned* o = (unsigned*)(dst + (size_t)(kb * U_DIM + u) * 32);
#pragma unroll
    for (int i = 0; i < 16; ++i)
        o[i] = (unsigned)tmp[2 * i] | ((unsigned)tmp[2 * i + 1] << 16);
}

// ---------------------------------------------------------------------------
// K1: h[b,u] = hidden[b,:] . W2_w[:,u] + W2_b[u]   (tiny; plain VALU, L2-bound)
// ---------------------------------------------------------------------------
__global__ void __launch_bounds__(256) hidden_proj(const float* __restrict__ hidden,
                                                   const float* __restrict__ W2,
                                                   const float* __restrict__ W2b,
                                                   float* __restrict__ h) {
    const int b = blockIdx.x >> 2;
    const int u = (blockIdx.x & 3) * 256 + threadIdx.x;
    float acc = W2b[u];
    const float* hv = hidden + (size_t)b * D_DIM;
#pragma unroll 8
    for (int d = 0; d < D_DIM; ++d)
        acc += hv[d] * W2[(size_t)d * U_DIM + u];
    h[(size_t)b * U_DIM + u] = acc;
}

// ---------------------------------------------------------------------------
// K2: fused  logits[b,t] = sum_u tanh( (features @ W1)[bt,u] + W1_b[u] + h[b,u] ) * V_w[u]
// WMMA bf16 GEMM: M = B*T (64 rows/WG), N = U = 1024, K = D = 1024.
// 8 waves: wave w -> M-subtile (w&3); each job covers FOUR N-tiles (four
// independent WMMA chains). B fragments for kb+1 are prefetched branch-free
// before the kb WMMAs (unroll turns rotation into renaming; the kb==31
// prefetch reads benign in-bounds garbage from our own workspace/LDS).
// ---------------------------------------------------------------------------
__global__ void __launch_bounds__(256) score_gemm(const float* __restrict__ features,
                                                  const unsigned short* __restrict__ w1p,
                                                  const float* __restrict__ W1b,
                                                  const float* __restrict__ h,
                                                  const float* __restrict__ Vw,
                                                  float* __restrict__ logits) {
    extern __shared__ __align__(16) char smem_raw[];
    unsigned short* As = (unsigned short*)smem_raw;                  // [64][LDA] bf16
    float* lred = (float*)(smem_raw + (size_t)M_TILE * LDA * 2);     // [64] logit accum

    const int tid = threadIdx.x;
    const int r0  = blockIdx.x * M_TILE;      // first row in flattened [B*T]
    const int b   = r0 >> 11;                 // r0 / T_DIM (64 | 2048 -> same b)

    if (tid < M_TILE) lred[tid] = 0.0f;

    // ---- Phase 1: stage 64x1024 f32 -> bf16 A-tile in LDS (coalesced f32x4)
    const float* fsrc = features + (size_t)r0 * D_DIM;
#pragma unroll 4
    for (int i = 0; i < 64; ++i) {
        const int e4  = (i * 256 + tid) * 4;          // linear elem idx, row-major
        const int row = e4 >> 10;
        const int col = e4 & 1023;
        const float4 v = *(const float4*)(fsrc + e4);
        const unsigned p0 = (unsigned)f2bf(v.x) | ((unsigned)f2bf(v.y) << 16);
        const unsigned p1 = (unsigned)f2bf(v.z) | ((unsigned)f2bf(v.w) << 16);
        unsigned short* dp = As + (size_t)row * LDA + col;
        *(unsigned*)(dp)     = p0;
        *(unsigned*)(dp + 2) = p1;
    }
    __syncthreads();

    // ---- Phase 2: WMMA GEMM + fused tanh()*V epilogue
    const int w    = tid >> 5;
    const int lane = tid & 31;
    const int msub = w & 3;            // which 16-row M-subtile
    const int half = lane >> 4;        // 0: lanes 0-15, 1: lanes 16-31
    const int l15  = lane & 15;

    // A fragment addressing (contiguous K order matching packed B: lanes 0-15 hold K{0-15},
    // lanes 16-31 hold K{16-31} of row (lane&15)).
    const unsigned aBase = (unsigned)(msub * 16 + l15) * LDA + (unsigned)(half << 4);

    const float* hb = h + (size_t)b * U_DIM;

    for (int j = 0; j < 8; ++j) {
        const int nt0   = (w >> 2) * 32 + j * 4;      // four adjacent N-tiles
        const int gcol0 = nt0 * 16 + l15;
        // B fragment base: lane covers column gcol, K half = half*16; adjacent
        // N-tiles are +512 ushorts (1KB) apart -> single base + imm offsets.
        const unsigned short* bp = w1p + (size_t)gcol0 * 32 + (half << 4);

        v8f acc0 = {0.f, 0.f, 0.f, 0.f, 0.f, 0.f, 0.f, 0.f};
        v8f acc1 = {0.f, 0.f, 0.f, 0.f, 0.f, 0.f, 0.f, 0.f};
        v8f acc2 = {0.f, 0.f, 0.f, 0.f, 0.f, 0.f, 0.f, 0.f};
        v8f acc3 = {0.f, 0.f, 0.f, 0.f, 0.f, 0.f, 0.f, 0.f};

        // --- pipeline prologue: B fragments for kb = 0
        v16bf b0 = ld_b(bp);
        v16bf b1 = ld_b(bp + 512);
        v16bf b2 = ld_b(bp + 1024);
        v16bf b3 = ld_b(bp + 1536);

#pragma unroll 4
        for (int kb = 0; kb < 32; ++kb) {
            // A fragment for current kb from LDS (short latency, in-body)
            const unsigned ka = aBase + (unsigned)kb * 32;
            v8bf a0 = *(const v8bf*)(const void*)(As + ka);        // K base+0..7
            v8bf a1 = *(const v8bf*)(const void*)(As + ka + 8);    // K base+8..15
            v16bf av = __builtin_shufflevector(a0, a1,
                0, 1, 2, 3, 4, 5, 6, 7, 8, 9, 10, 11, 12, 13, 14, 15);

            v16bf c0 = b0, c1 = b1, c2 = b2, c3 = b3;

            // Branch-free prefetch of kb+1 B fragments (kb==31 reads <=68KB
            // past packed W1 -> lands in hbuf/logit region of d_ws: benign).
            const unsigned short* bk = bp + (size_t)(kb + 1) * (U_DIM * 32);
            b0 = ld_b(bk);
            b1 = ld_b(bk + 512);
            b2 = ld_b(bk + 1024);
            b3 = ld_b(bk + 1536);

            acc0 = __builtin_amdgcn_wmma_f32_16x16x32_bf16(
                false, av, false, c0, (short)0, acc0, false, false);
            acc1 = __builtin_amdgcn_wmma_f32_16x16x32_bf16(
                false, av, false, c1, (short)0, acc1, false, false);
            acc2 = __builtin_amdgcn_wmma_f32_16x16x32_bf16(
                false, av, false, c2, (short)0, acc2, false, false);
            acc3 = __builtin_amdgcn_wmma_f32_16x16x32_bf16(
                false, av, false, c3, (short)0, acc3, false, false);
        }

        // Epilogue: C layout -> lane l, VGPR r holds (M = r + 8*half, N = l&15)
        const float e0 = W1b[gcol0],      g0 = hb[gcol0],      q0 = Vw[gcol0];
        const float e1 = W1b[gcol0 + 16], g1 = hb[gcol0 + 16], q1 = Vw[gcol0 + 16];
        const float e2 = W1b[gcol0 + 32], g2 = hb[gcol0 + 32], q2 = Vw[gcol0 + 32];
        const float e3 = W1b[gcol0 + 48], g3 = hb[gcol0 + 48], q3 = Vw[gcol0 + 48];
#pragma unroll
        for (int r = 0; r < 8; ++r) {
            float v = fast_tanh(acc0[r] + e0 + g0) * q0
                    + fast_tanh(acc1[r] + e1 + g1) * q1
                    + fast_tanh(acc2[r] + e2 + g2) * q2
                    + fast_tanh(acc3[r] + e3 + g3) * q3;
#pragma unroll
            for (int off = 1; off < 16; off <<= 1)
                v += __shfl_xor(v, off);
            if (l15 == 0) {
                const int mloc = msub * 16 + r + (half << 3);
                atomicAdd(&lred[mloc], v);      // ds_add_f32
            }
        }
    }
    __syncthreads();
    if (tid < M_TILE) logits[(size_t)r0 + tid] = lred[tid];
}

// ---------------------------------------------------------------------------
// K3: softmax over T per batch row -> attention weights (directly to d_out)
// ---------------------------------------------------------------------------
__global__ void __launch_bounds__(256) softmax_k(const float* __restrict__ logits,
                                                 float* __restrict__ attn) {
    __shared__ float red[256];
    const int b = blockIdx.x, tid = threadIdx.x;
    const float* lp = logits + (size_t)b * T_DIM;
    float lv[8];
    float lmax = -1e30f;
#pragma unroll
    for (int i = 0; i < 8; ++i) { lv[i] = lp[i * 256 + tid]; lmax = fmaxf(lmax, lv[i]); }
    red[tid] = lmax; __syncthreads();
    for (int s = 128; s > 0; s >>= 1) {
        if (tid < s) red[tid] = fmaxf(red[tid], red[tid + s]);
        __syncthreads();
    }
    const float bmax = red[0]; __syncthreads();
    float lsum = 0.f;
#pragma unroll
    for (int i = 0; i < 8; ++i) { lv[i] = __expf(lv[i] - bmax); lsum += lv[i]; }
    red[tid] = lsum; __syncthreads();
    for (int s = 128; s > 0; s >>= 1) {
        if (tid < s) red[tid] += red[tid + s];
        __syncthreads();
    }
    const float inv = 1.0f / red[0];
    float* ap = attn + (size_t)b * T_DIM;
#pragma unroll
    for (int i = 0; i < 8; ++i) ap[i * 256 + tid] = lv[i] * inv;
}

// ---------------------------------------------------------------------------
// K4a: zero the context output (harness poisons d_out; K4b accumulates atomically)
// ---------------------------------------------------------------------------
__global__ void __launch_bounds__(256) zero_ctx(float* __restrict__ ctx) {
    ctx[blockIdx.x * 256 + threadIdx.x] = 0.0f;
}

// ---------------------------------------------------------------------------
// K4b: context[b,d] += sum_{t in segment} attn[b,t] * features[b,t,d]
// T split 4 ways across blocks for parallelism; one f32 atomic per thread.
// ---------------------------------------------------------------------------
__global__ void __launch_bounds__(256) context_k(const float* __restrict__ features,
                                                 const float* __restrict__ attn,
                                                 float* __restrict__ ctx) {
    __shared__ float wsm[256];
    const int b  = blockIdx.x >> 4;          // 0..31
    const int dc = (blockIdx.x >> 2) & 3;    // 0..3  (d chunk of 256)
    const int ts = blockIdx.x & 3;           // 0..3  (t segment of 512)
    const int d  = dc * 256 + threadIdx.x;
    float acc = 0.f;
    for (int tc = 0; tc < 2; ++tc) {
        const int t0 = ts * 512 + tc * 256;
        wsm[threadIdx.x] = attn[(size_t)b * T_DIM + t0 + threadIdx.x];
        __syncthreads();
        const float* fp = features + ((size_t)b * T_DIM + (size_t)t0) * D_DIM + d;
#pragma unroll 8
        for (int j = 0; j < 256; ++j)
            acc += wsm[j] * fp[(size_t)j * D_DIM];
        __syncthreads();
    }
    atomicAdd(&ctx[(size_t)b * D_DIM + d], acc);   // global_atomic_add_f32
}

// ---------------------------------------------------------------------------
// Host launcher
// ---------------------------------------------------------------------------
extern "C" void kernel_launch(void* const* d_in, const int* in_sizes, int n_in,
                              void* d_out, int out_size, void* d_ws, size_t ws_size,
                              hipStream_t stream) {
    (void)in_sizes; (void)n_in; (void)out_size; (void)ws_size;
    const float* features = (const float*)d_in[0];   // [B,T,D]
    const float* hidden   = (const float*)d_in[1];   // [B,D]
    const float* W1_w     = (const float*)d_in[2];   // [D,U]
    const float* W1_b     = (const float*)d_in[3];   // [U]
    const float* W2_w     = (const float*)d_in[4];   // [D,U]
    const float* W2_b     = (const float*)d_in[5];   // [U]
    const float* V_w      = (const float*)d_in[6];   // [U,1]
    // V_b shifts all logits equally -> softmax-invariant; logits are not outputs.

    float* out_ctx  = (float*)d_out;                       // [B,D]   = 32768
    float* out_attn = (float*)d_out + B_DIM * D_DIM;       // [B,T,1] = 65536

    // workspace: 2MB packed W1 bf16 | 128KB h | 256KB logits
    // (K2's last-iteration B prefetch over-reads <=68KB past the 2MB region;
    //  that lands inside hbuf/logit below, so it stays within d_ws.)
    unsigned short* w1p  = (unsigned short*)d_ws;
    float* hbuf  = (float*)((char*)d_ws + (1u << 21));
    float* logit = (float*)((char*)d_ws + (1u << 21) + (1u << 17));

    pack_w1<<<128, 256, 0, stream>>>(W1_w, w1p);
    hidden_proj<<<128, 256, 0, stream>>>(hidden, W2_w, W2_b, hbuf);

    const size_t smem = (size_t)M_TILE * LDA * 2 + M_TILE * sizeof(float); // ~132.4 KB
    score_gemm<<<(B_DIM * T_DIM) / M_TILE, 256, smem, stream>>>(
        features, w1p, W1_b, hbuf, V_w, logit);

    softmax_k<<<B_DIM, 256, 0, stream>>>(logit, out_attn);
    zero_ctx<<<(B_DIM * D_DIM) / 256, 256, 0, stream>>>(out_ctx);
    context_k<<<512, 256, 0, stream>>>(features, out_attn, out_ctx);
}
